// CausalSelfAttention_45346264711572
// MI455X (gfx1250) — matrix operand order repaired and checked by reading, not compiled, not run on hardware
//
#include <hip/hip_runtime.h>
#include <hip/hip_bf16.h>
#include <math.h>

// ---------------- constants (problem shape) ----------------
#define BSZ   4
#define TLEN  2048
#define CDIM  768
#define HEADS 12
#define HD    64
#define C3    (3 * CDIM)      // 2304
#define MROWS (BSZ * TLEN)    // 8192

typedef __attribute__((ext_vector_type(8)))  _Float16 v8h;
typedef __attribute__((ext_vector_type(16))) _Float16 v16h;
typedef __attribute__((ext_vector_type(8)))  float    v8f;

static __device__ __forceinline__ v8f vzero8f() {
  v8f z;
#pragma unroll
  for (int i = 0; i < 8; ++i) z[i] = 0.0f;
  return z;
}

// WMMA D = A(16x32 f16) * B(32x16 f16) + C(16x16 f32)
static __device__ __forceinline__ v8f wmma_f16(v16h a, v16h b, v8f c) {
  return __builtin_amdgcn_wmma_f32_16x16x32_f16(false, a, false, b, (short)0, c,
                                                false, false);
}

// ---- A operand load: 16x32 f16 tile, row-major, stride lda (elements).
// Lane L: m = L%16, half = L/16.
//   elements 0..7  : k = half*8 + e        (contiguous 16B)
//   elements 8..15 : k = 16 + half*8 + e   (contiguous 16B)
static __device__ __forceinline__ v16h load_A16x32(const _Float16* Abase, int lda) {
  int lane = threadIdx.x & 31;
  int m = lane & 15, hf = lane >> 4;
  const _Float16* p = Abase + m * lda + hf * 8;
  v8h lo = *(const v8h*)(p);
  v8h hi = *(const v8h*)(p + 16);
  v16h r;
#pragma unroll
  for (int i = 0; i < 8; ++i) { r[i] = lo[i]; r[8 + i] = hi[i]; }
  return r;
}

// ---- B operand load: 32x16 f16 tile B[k][n], read from TRANSPOSED storage
// Bt[n][k] (row-major, stride ldb). Lane L: n = L%16, half = L/16,
// elements 0..7 -> k = half*8 + e, elements 8..15 -> k = 16 + half*8 + e.
static __device__ __forceinline__ v16h load_B32x16T(const _Float16* Bt, int ldb) {
  int lane = threadIdx.x & 31;
  int n = lane & 15, hf = lane >> 4;
  const _Float16* p = Bt + n * ldb + hf * 8;
  v8h lo = *(const v8h*)(p);
  v8h hi = *(const v8h*)(p + 16);
  v16h r;
#pragma unroll
  for (int i = 0; i < 8; ++i) { r[i] = lo[i]; r[8 + i] = hi[i]; }
  return r;
}

// ---------------- conversion / transpose kernels ----------------
__global__ void cvt_f32_to_f16(const float* __restrict__ src,
                               _Float16* __restrict__ dst, int n) {
  int i = blockIdx.x * blockDim.x + threadIdx.x;
  if (i < n) dst[i] = (_Float16)src[i];
}

// W[rows][cols] (f32) -> Wt[cols][rows] (f16)
__global__ void transpose_cvt(const float* __restrict__ W,
                              _Float16* __restrict__ Wt, int rows, int cols) {
  int i = blockIdx.x * blockDim.x + threadIdx.x;
  if (i < rows * cols) {
    int r = i / cols, c = i - r * cols;
    Wt[(size_t)c * rows + r] = (_Float16)W[i];
  }
}

// ---------------- QKV GEMM: [8192x768] x [768x2304] + bias ----------------
// One wave computes a 32x32 tile. Epilogue scatters into Q[b,h,t,d] (scaled
// by 1/sqrt(hd)), K[b,h,t,d], and transposed Vt[b,h,d,t], all f16.
__global__ __launch_bounds__(256) void qkv_gemm(
    const _Float16* __restrict__ Xh, const _Float16* __restrict__ WaT,
    const float* __restrict__ b_attn, _Float16* __restrict__ Qh,
    _Float16* __restrict__ Kh, _Float16* __restrict__ Vt) {
  const int Ntiles = C3 / 32;  // 72
  int wave = (blockIdx.x * blockDim.x + threadIdx.x) >> 5;
  int mb = wave / Ntiles, nb = wave - mb * Ntiles;
  if (mb >= MROWS / 32) return;
  int m0 = mb * 32, n0 = nb * 32;

  v8f acc00 = vzero8f(), acc01 = vzero8f(), acc10 = vzero8f(), acc11 = vzero8f();
  for (int k = 0; k < CDIM; k += 32) {
    v16h a0 = load_A16x32(Xh + (size_t)m0 * CDIM + k, CDIM);
    v16h a1 = load_A16x32(Xh + (size_t)(m0 + 16) * CDIM + k, CDIM);
    v16h b0 = load_B32x16T(WaT + (size_t)n0 * CDIM + k, CDIM);
    v16h b1 = load_B32x16T(WaT + (size_t)(n0 + 16) * CDIM + k, CDIM);
    acc00 = wmma_f16(a0, b0, acc00);
    acc01 = wmma_f16(a0, b1, acc01);
    acc10 = wmma_f16(a1, b0, acc10);
    acc11 = wmma_f16(a1, b1, acc11);
  }

  int lane = threadIdx.x & 31, nL = lane & 15, hf = lane >> 4;
  const float qscale = 0.125f;  // 1/sqrt(64)
#pragma unroll
  for (int i = 0; i < 2; ++i) {
#pragma unroll
    for (int j = 0; j < 2; ++j) {
      v8f acc = (i == 0) ? (j == 0 ? acc00 : acc01) : (j == 0 ? acc10 : acc11);
#pragma unroll
      for (int r = 0; r < 8; ++r) {
        int row = m0 + i * 16 + r + 8 * hf;          // 0..8191
        int col = n0 + j * 16 + nL;                  // 0..2303
        float v = acc[r] + b_attn[col];
        int seg = col / CDIM;
        int cc = col - seg * CDIM;
        int h = cc >> 6, d = cc & 63;
        int b = row >> 11, t = row & (TLEN - 1);
        size_t bh = (size_t)b * HEADS + h;
        if (seg == 0) {
          Qh[(bh * TLEN + t) * HD + d] = (_Float16)(v * qscale);
        } else if (seg == 1) {
          Kh[(bh * TLEN + t) * HD + d] = (_Float16)v;
        } else {
          Vt[(bh * HD + d) * TLEN + t] = (_Float16)v;
        }
      }
    }
  }
}

// ---------------- flash attention: one wave per (b,h, 16-query tile) -------
__global__ __launch_bounds__(128) void attn_kernel(
    const _Float16* __restrict__ Qh, const _Float16* __restrict__ Kh,
    const _Float16* __restrict__ Vt, _Float16* __restrict__ Yh) {
  __shared__ __align__(16) _Float16 Pbuf[4][16 * 32];  // per-wave P staging

  int wavel = threadIdx.x >> 5;
  int wave = blockIdx.x * 4 + wavel;
  int qt = wave & 127;          // T/16 = 128 query tiles
  int bh = wave >> 7;           // 0..47
  if (bh >= BSZ * HEADS) return;
  int q0 = qt * 16;

  const _Float16* Qb = Qh + ((size_t)bh * TLEN + q0) * HD;
  const _Float16* Kb = Kh + (size_t)bh * TLEN * HD;
  const _Float16* Vb = Vt + (size_t)bh * HD * TLEN;

  // Q tile (16x64) in registers as two A operands (WMMA-k = d)
  v16h aq0 = load_A16x32(Qb, HD);       // d 0..31
  v16h aq1 = load_A16x32(Qb + 32, HD);  // d 32..63

  v8f o0 = vzero8f(), o1 = vzero8f(), o2 = vzero8f(), o3 = vzero8f();
  float mi[8], li[8];
#pragma unroll
  for (int r = 0; r < 8; ++r) { mi[r] = -INFINITY; li[r] = 0.0f; }

  int lane = threadIdx.x & 31, nL = lane & 15, hf = lane >> 4;
  _Float16* P = Pbuf[wavel];

  for (int kb = 0; kb < q0 + 16; kb += 32) {
    // ---- S = Q K^T for 32 keys (two 16x16 C tiles) ----
    const _Float16* Krow = Kb + (size_t)kb * HD;
    v16h bk0a = load_B32x16T(Krow, HD);             // keys kb..kb+15, d 0..31
    v16h bk0b = load_B32x16T(Krow + 32, HD);        // d 32..63
    v16h bk1a = load_B32x16T(Krow + 16 * HD, HD);   // keys kb+16..kb+31
    v16h bk1b = load_B32x16T(Krow + 16 * HD + 32, HD);
    v8f s0 = vzero8f(), s1 = vzero8f();
    s0 = wmma_f16(aq0, bk0a, s0);
    s0 = wmma_f16(aq1, bk0b, s0);
    s1 = wmma_f16(aq0, bk1a, s1);
    s1 = wmma_f16(aq1, bk1b, s1);

    // ---- causal mask + online softmax (row M = r + 8*hf, col N = nL) ----
#pragma unroll
    for (int r = 0; r < 8; ++r) {
      int qrow = q0 + r + 8 * hf;
      float x0 = (kb + nL <= qrow) ? s0[r] : -INFINITY;
      float x1 = (kb + 16 + nL <= qrow) ? s1[r] : -INFINITY;
      float mx = fmaxf(x0, x1);
#pragma unroll
      for (int off = 8; off; off >>= 1) mx = fmaxf(mx, __shfl_xor(mx, off, 32));
      float mnew = fmaxf(mi[r], mx);
      float p0 = __expf(x0 - mnew);
      float p1 = __expf(x1 - mnew);
      float rs = p0 + p1;
#pragma unroll
      for (int off = 8; off; off >>= 1) rs += __shfl_xor(rs, off, 32);
      float corr = __expf(mi[r] - mnew);
      li[r] = li[r] * corr + rs;
      mi[r] = mnew;
      o0[r] *= corr; o1[r] *= corr; o2[r] *= corr; o3[r] *= corr;
      int row = r + 8 * hf;
      P[row * 32 + nL] = (_Float16)p0;
      P[row * 32 + 16 + nL] = (_Float16)p1;
    }
    // whole-wave LDS store->load round trip (C-layout -> A-layout transpose)
    asm volatile("s_wait_dscnt 0" ::: "memory");

    // ---- O += P V  (A = P 16x32, B = V 32x16 per d-block, from Vt) ----
    v16h ap = load_A16x32(P, 32);
    v16h bv0 = load_B32x16T(Vb + (size_t)0 * 16 * TLEN + kb, TLEN);
    v16h bv1 = load_B32x16T(Vb + (size_t)1 * 16 * TLEN + kb, TLEN);
    v16h bv2 = load_B32x16T(Vb + (size_t)2 * 16 * TLEN + kb, TLEN);
    v16h bv3 = load_B32x16T(Vb + (size_t)3 * 16 * TLEN + kb, TLEN);
    o0 = wmma_f16(ap, bv0, o0);
    o1 = wmma_f16(ap, bv1, o1);
    o2 = wmma_f16(ap, bv2, o2);
    o3 = wmma_f16(ap, bv3, o3);
  }

  // ---- epilogue: O /= l, write Y[b,t, h*64 + d] as f16 ----
  int b = bh / HEADS, h = bh - b * HEADS;
#pragma unroll
  for (int r = 0; r < 8; ++r) {
    int t = q0 + r + 8 * hf;
    float inv = 1.0f / li[r];
    size_t base = ((size_t)b * TLEN + t) * CDIM + h * HD;
    Yh[base + 0 * 16 + nL] = (_Float16)(o0[r] * inv);
    Yh[base + 1 * 16 + nL] = (_Float16)(o1[r] * inv);
    Yh[base + 2 * 16 + nL] = (_Float16)(o2[r] * inv);
    Yh[base + 3 * 16 + nL] = (_Float16)(o3[r] * inv);
  }
}

// ---------------- projection GEMM: [8192x768] x [768x768] + bias -> f32 ----
__global__ __launch_bounds__(256) void proj_gemm(
    const _Float16* __restrict__ Yh, const _Float16* __restrict__ WpT,
    const float* __restrict__ b_proj, float* __restrict__ out) {
  const int Ntiles = CDIM / 32;  // 24
  int wave = (blockIdx.x * blockDim.x + threadIdx.x) >> 5;
  int mb = wave / Ntiles, nb = wave - mb * Ntiles;
  if (mb >= MROWS / 32) return;
  int m0 = mb * 32, n0 = nb * 32;

  v8f acc00 = vzero8f(), acc01 = vzero8f(), acc10 = vzero8f(), acc11 = vzero8f();
  for (int k = 0; k < CDIM; k += 32) {
    v16h a0 = load_A16x32(Yh + (size_t)m0 * CDIM + k, CDIM);
    v16h a1 = load_A16x32(Yh + (size_t)(m0 + 16) * CDIM + k, CDIM);
    v16h b0 = load_B32x16T(WpT + (size_t)n0 * CDIM + k, CDIM);
    v16h b1 = load_B32x16T(WpT + (size_t)(n0 + 16) * CDIM + k, CDIM);
    acc00 = wmma_f16(a0, b0, acc00);
    acc01 = wmma_f16(a0, b1, acc01);
    acc10 = wmma_f16(a1, b0, acc10);
    acc11 = wmma_f16(a1, b1, acc11);
  }

  int lane = threadIdx.x & 31, nL = lane & 15, hf = lane >> 4;
#pragma unroll
  for (int i = 0; i < 2; ++i) {
#pragma unroll
    for (int j = 0; j < 2; ++j) {
      v8f acc = (i == 0) ? (j == 0 ? acc00 : acc01) : (j == 0 ? acc10 : acc11);
#pragma unroll
      for (int r = 0; r < 8; ++r) {
        int row = m0 + i * 16 + r + 8 * hf;
        int col = n0 + j * 16 + nL;
        out[(size_t)row * CDIM + col] = acc[r] + b_proj[col];
      }
    }
  }
}

// ---------------- host launcher ----------------
extern "C" void kernel_launch(void* const* d_in, const int* in_sizes, int n_in,
                              void* d_out, int out_size, void* d_ws, size_t ws_size,
                              hipStream_t stream) {
  const float* x      = (const float*)d_in[0];  // [4,2048,768]
  const float* W_attn = (const float*)d_in[1];  // [768,2304]
  const float* b_attn = (const float*)d_in[2];  // [2304]
  const float* W_proj = (const float*)d_in[3];  // [768,768]
  const float* b_proj = (const float*)d_in[4];  // [768]
  float* out = (float*)d_out;

  // workspace carve-up (256B aligned)
  char* ws = (char*)d_ws;
  size_t off = 0;
  auto carve = [&](size_t bytes) -> char* {
    char* p = ws + off;
    off += (bytes + 255) & ~(size_t)255;
    return p;
  };
  _Float16* Xh  = (_Float16*)carve((size_t)MROWS * CDIM * 2);   // 12.6 MB
  _Float16* WaT = (_Float16*)carve((size_t)C3 * CDIM * 2);      //  3.5 MB
  _Float16* WpT = (_Float16*)carve((size_t)CDIM * CDIM * 2);    //  1.2 MB
  _Float16* Qh  = (_Float16*)carve((size_t)BSZ * HEADS * TLEN * HD * 2);
  _Float16* Kh  = (_Float16*)carve((size_t)BSZ * HEADS * TLEN * HD * 2);
  _Float16* Vt  = (_Float16*)carve((size_t)BSZ * HEADS * HD * TLEN * 2);
  _Float16* Yh  = (_Float16*)carve((size_t)MROWS * CDIM * 2);
  (void)ws_size;

  // 1) precision / layout conversions
  {
    int n = MROWS * CDIM;
    cvt_f32_to_f16<<<(n + 255) / 256, 256, 0, stream>>>(x, Xh, n);
  }
  {
    int n = CDIM * C3;
    transpose_cvt<<<(n + 255) / 256, 256, 0, stream>>>(W_attn, WaT, CDIM, C3);
  }
  {
    int n = CDIM * CDIM;
    transpose_cvt<<<(n + 255) / 256, 256, 0, stream>>>(W_proj, WpT, CDIM, CDIM);
  }

  // 2) QKV GEMM: (8192/32)*(2304/32) = 18432 waves, 8 waves/block
  {
    int waves = (MROWS / 32) * (C3 / 32);
    qkv_gemm<<<waves / 8, 256, 0, stream>>>(Xh, WaT, b_attn, Qh, Kh, Vt);
  }

  // 3) attention: B*H*(T/16) = 6144 waves, 4 waves/block
  {
    int waves = BSZ * HEADS * (TLEN / 16);
    attn_kernel<<<waves / 4, 128, 0, stream>>>(Qh, Kh, Vt, Yh);
  }

  // 4) projection GEMM: (8192/32)*(768/32) = 6144 waves, 8 waves/block
  {
    int waves = (MROWS / 32) * (CDIM / 32);
    proj_gemm<<<waves / 8, 256, 0, stream>>>(Yh, WpT, b_proj, out);
  }
}
